// OFTv2Module_42717744726325
// MI455X (gfx1250) — matrix operand order repaired and checked by reading, not compiled, not run on hardware
//
#include <hip/hip_runtime.h>

#define IN_F  4096
#define OUT_F 4096
#define RANK  128
#define M_TOK 8192   // 4 * 2048
#define NUM_TERMS 5

typedef __attribute__((ext_vector_type(16))) __bf16 bfx16;
typedef __attribute__((ext_vector_type(8)))  __bf16 bfx8;
typedef __attribute__((ext_vector_type(8)))  float  fx8;

__device__ __forceinline__ unsigned short f32_to_bf16_rne(float f) {
    unsigned u = __float_as_uint(f);
    unsigned r = u + 0x7FFFu + ((u >> 16) & 1u);
    return (unsigned short)(r >> 16);
}
__device__ __forceinline__ float bf16_to_f32(unsigned short h) {
    return __uint_as_float(((unsigned)h) << 16);
}

// ---------------------------------------------------------------------------
// Kernel 1: R[r] = I + 2Q + 2Q^2 + 2Q^3 + 2Q^4  (Cayley-Neumann, 5 terms)
// ---------------------------------------------------------------------------
__global__ void oft_build_R(const float* __restrict__ vec, float* __restrict__ Rout) {
    __shared__ float Q[32][33];
    __shared__ float P[32][33];
    const int j = threadIdx.x & 31;
    const int i = threadIdx.x >> 5;
    const int r = blockIdx.x;
    float q = 0.f;
    if (i < j) q =  vec[r * 496 + (i * (63 - i)) / 2 + (j - i - 1)];
    if (j < i) q = -vec[r * 496 + (j * (63 - j)) / 2 + (i - j - 1)];
    Q[i][j] = q;
    P[i][j] = q;
    float res = ((i == j) ? 1.f : 0.f) + 2.f * q;
    __syncthreads();
    for (int t = 2; t < NUM_TERMS; ++t) {
        float acc = 0.f;
        #pragma unroll
        for (int k = 0; k < 32; ++k) acc += P[i][k] * Q[k][j];
        __syncthreads();
        P[i][j] = acc;
        res += 2.f * acc;
        __syncthreads();
    }
    Rout[r * 1024 + i * 32 + j] = res;
}

// ---------------------------------------------------------------------------
// Kernel 2: fold rotation into frozen weight. W'[o, r*32+k] = sum_c W[o,r*32+c]*R[r,k,c]
// Split result into bf16 hi + bf16 lo (residual) for the split-precision GEMM.
// ---------------------------------------------------------------------------
__global__ void oft_fold_W(const float* __restrict__ W, const float* __restrict__ R,
                           unsigned short* __restrict__ wph, unsigned short* __restrict__ wpl) {
    __shared__ float Rl[32][33];
    __shared__ float Wl[8][32];
    const int t  = threadIdx.x;
    const int k  = t & 31;
    const int ol = t >> 5;                 // 0..7
    const int r  = blockIdx.y;
    const int o  = blockIdx.x * 8 + ol;
    #pragma unroll
    for (int e = 0; e < 4; ++e) {
        int idx = t + e * 256;
        Rl[idx >> 5][idx & 31] = R[r * 1024 + idx];
    }
    Wl[ol][k] = W[(size_t)o * IN_F + r * 32 + k];
    __syncthreads();
    float acc = 0.f;
    #pragma unroll
    for (int c = 0; c < 32; ++c) acc += Wl[ol][c] * Rl[k][c];
    unsigned short hb = f32_to_bf16_rne(acc);
    float lo = acc - bf16_to_f32(hb);
    size_t oi = (size_t)o * IN_F + r * 32 + k;
    wph[oi] = hb;
    wpl[oi] = f32_to_bf16_rne(lo);
}

// ---------------------------------------------------------------------------
// TDM helper: async DMA of one 128x32 bf16 tile (row-major [N,K], stride 4096)
// into LDS with hardware padding to a 96-byte row stride.
//   pad_interval=3 -> pad after every 16 DWORDs (= one 64B tile row)
//   pad_amount=7   -> insert 8 DWORDs (32B)  => LDS row stride 96B == LDT*2
// ---------------------------------------------------------------------------
#if defined(__has_builtin)
#if __has_builtin(__builtin_amdgcn_tensor_load_to_lds)
#define HAVE_TDM 1
#endif
#endif

#ifdef HAVE_TDM
typedef __attribute__((ext_vector_type(4))) unsigned int tdm_g0_t;
typedef __attribute__((ext_vector_type(8))) int          tdm_g1_t;
typedef __attribute__((ext_vector_type(4))) int          tdm_g2_t;
typedef __attribute__((ext_vector_type(8))) int          tdm_g4_t;

__device__ __forceinline__ void tdm_load_tile_bf16(unsigned lds_byte_off,
                                                   const unsigned short* gptr) {
    unsigned long long ga = (unsigned long long)(size_t)gptr;
    tdm_g0_t g0;
    g0.x = 1u;                                                  // count=1, user D#
    g0.y = lds_byte_off;                                        // lds_addr
    g0.z = (unsigned)ga;                                        // global_addr[31:0]
    g0.w = (unsigned)((ga >> 32) & 0x01FFFFFFu) | 0x80000000u;  // addr[56:32] | type=2
    tdm_g1_t g1;
    g1[0] = (1 << 16)     // data_size = 2 bytes
          | (1 << 20)     // pad_enable
          | (3 << 22)     // pad_interval = 16 DWORDs
          | (7 << 25);    // pad_amount   = 8 DWORDs
    g1[1] = (int)((4096u & 0xFFFFu) << 16);                     // tensor_dim0 lo
    g1[2] = (int)(((4096u >> 16) & 0xFFFFu) | ((4096u & 0xFFFFu) << 16)); // dim0 hi | dim1 lo
    g1[3] = (int)(((4096u >> 16) & 0xFFFFu) | (32u << 16));     // dim1 hi | tile_dim0=32
    g1[4] = 128;                                                // tile_dim1=128, tile_dim2=0
    g1[5] = 4096;                                               // tensor_dim0_stride[31:0]
    g1[6] = 0;                                                  // stride hi / dim1_stride
    g1[7] = 0;
    tdm_g2_t gz  = {0, 0, 0, 0};                                // groups 2/3 unused (2-D)
    tdm_g4_t gz8 = {0, 0, 0, 0, 0, 0, 0, 0};                    // extra group (clang-23 form)
    __builtin_amdgcn_tensor_load_to_lds(g0, g1, gz, gz, gz8, 0);
}
#endif

// ---------------------------------------------------------------------------
// Kernel 3: out[M,N] = x[M,K] @ W'^T + bias, split-precision bf16x2 WMMA GEMM
// 128x128x32 workgroup tile, 8 waves, each wave 64x32 = 4x2 accumulators.
// B tiles staged by the Tensor Data Mover; A tile split f32->bf16 hi/lo by VALU
// (co-executes with the XDL WMMA pipe).
// ---------------------------------------------------------------------------
#define BM 128
#define BN 128
#define BK 32
#define LDT 48   // padded LDS row stride in bf16 elements (96 bytes, 16B aligned)

__global__ __launch_bounds__(256)
void oft_gemm(const float* __restrict__ x,
              const unsigned short* __restrict__ wph,
              const unsigned short* __restrict__ wpl,
              const float* __restrict__ bias,
              float* __restrict__ out) {
    __shared__ unsigned short Ah[BM][LDT];
    __shared__ unsigned short Al[BM][LDT];
    __shared__ unsigned short Bh[BN][LDT];
    __shared__ unsigned short Bl[BN][LDT];

    const int t    = threadIdx.x;
    const int lane = t & 31;
    const int h    = lane >> 4;   // lane-half selects K sub-ranges
    const int lm   = lane & 15;
    const int wave = t >> 5;      // 0..7
    const int wm   = wave & 1;    // M direction (2 waves)
    const int wn   = wave >> 1;   // N direction (4 waves)

    const int m0 = blockIdx.y * BM;
    const int n0 = blockIdx.x * BN;

    fx8 acc[4][2];
    #pragma unroll
    for (int ms = 0; ms < 4; ++ms)
        #pragma unroll
        for (int ns = 0; ns < 2; ++ns)
            #pragma unroll
            for (int j = 0; j < 8; ++j) acc[ms][ns][j] = 0.f;

    for (int k0 = 0; k0 < IN_F; k0 += BK) {
        __syncthreads();   // previous iteration's LDS reads complete

#ifdef HAVE_TDM
        if (wave == 0) {
            tdm_load_tile_bf16((unsigned)(size_t)&Bh[0][0],
                               wph + (size_t)n0 * IN_F + k0);
            tdm_load_tile_bf16((unsigned)(size_t)&Bl[0][0],
                               wpl + (size_t)n0 * IN_F + k0);
        }
#else
        #pragma unroll
        for (int p = 0; p < 2; ++p) {
            int idx = p * 256 + t;
            int row = idx >> 2;
            int q   = idx & 3;
            size_t off = (size_t)(n0 + row) * IN_F + k0 + q * 8;
            *reinterpret_cast<uint4*>(&Bh[row][q * 8]) =
                *reinterpret_cast<const uint4*>(wph + off);
            *reinterpret_cast<uint4*>(&Bl[row][q * 8]) =
                *reinterpret_cast<const uint4*>(wpl + off);
        }
#endif

        // --- stage A tile (128x32 f32), split into bf16 hi/lo on the fly ---
        #pragma unroll
        for (int p = 0; p < 4; ++p) {
            int row = p * 32 + (t >> 3);
            int c4  = (t & 7) * 4;
            const float4 v = *reinterpret_cast<const float4*>(
                x + (size_t)(m0 + row) * IN_F + k0 + c4);
            float vv[4] = {v.x, v.y, v.z, v.w};
            #pragma unroll
            for (int e = 0; e < 4; ++e) {
                unsigned short hb = f32_to_bf16_rne(vv[e]);
                Ah[row][c4 + e] = hb;
                Al[row][c4 + e] = f32_to_bf16_rne(vv[e] - bf16_to_f32(hb));
            }
        }

#ifdef HAVE_TDM
        if (wave == 0) __builtin_amdgcn_s_wait_tensorcnt(0);  // B tiles landed
#endif
        __syncthreads();

        // --- fragments: A lane = 16 contiguous K, B lane = 16 contiguous K ---
        bfx16 a_h[4], a_l[4];
        #pragma unroll
        for (int ms = 0; ms < 4; ++ms) {
            const unsigned short* rp = &Ah[wm * 64 + ms * 16 + lm][0];
            bfx8 x0 = *reinterpret_cast<const bfx8*>(rp + 16 * h);
            bfx8 x1 = *reinterpret_cast<const bfx8*>(rp + 16 * h + 8);
            a_h[ms] = __builtin_shufflevector(x0, x1, 0,1,2,3,4,5,6,7,8,9,10,11,12,13,14,15);
            const unsigned short* rq = &Al[wm * 64 + ms * 16 + lm][0];
            bfx8 y0 = *reinterpret_cast<const bfx8*>(rq + 16 * h);
            bfx8 y1 = *reinterpret_cast<const bfx8*>(rq + 16 * h + 8);
            a_l[ms] = __builtin_shufflevector(y0, y1, 0,1,2,3,4,5,6,7,8,9,10,11,12,13,14,15);
        }
        bfx16 b_h[2], b_l[2];
        #pragma unroll
        for (int ns = 0; ns < 2; ++ns) {
            b_h[ns] = *reinterpret_cast<const bfx16*>(&Bh[wn * 32 + ns * 16 + lm][16 * h]);
            b_l[ns] = *reinterpret_cast<const bfx16*>(&Bl[wn * 32 + ns * 16 + lm][16 * h]);
        }

        #pragma unroll
        for (int ms = 0; ms < 4; ++ms)
            #pragma unroll
            for (int ns = 0; ns < 2; ++ns) {
                acc[ms][ns] = __builtin_amdgcn_wmma_f32_16x16x32_bf16(
                    false, a_h[ms], false, b_h[ns], (short)0, acc[ms][ns], false, false);
                acc[ms][ns] = __builtin_amdgcn_wmma_f32_16x16x32_bf16(
                    false, a_h[ms], false, b_l[ns], (short)0, acc[ms][ns], false, false);
                acc[ms][ns] = __builtin_amdgcn_wmma_f32_16x16x32_bf16(
                    false, a_l[ms], false, b_h[ns], (short)0, acc[ms][ns], false, false);
            }
    }

    // --- epilogue: D layout (VGPR j -> row j + 8h, col = lane%16), add bias ---
    #pragma unroll
    for (int ms = 0; ms < 4; ++ms) {
        #pragma unroll
        for (int ns = 0; ns < 2; ++ns) {
            int col = n0 + wn * 32 + ns * 16 + lm;
            float bv = bias[col];
            int rbase = m0 + wm * 64 + ms * 16 + 8 * h;
            #pragma unroll
            for (int j = 0; j < 8; ++j)
                out[(size_t)(rbase + j) * OUT_F + col] = acc[ms][ns][j] + bv;
        }
    }
}

// ---------------------------------------------------------------------------
extern "C" void kernel_launch(void* const* d_in, const int* in_sizes, int n_in,
                              void* d_out, int out_size, void* d_ws, size_t ws_size,
                              hipStream_t stream) {
    const float* x    = (const float*)d_in[0];
    const float* oftw = (const float*)d_in[1];
    const float* W    = (const float*)d_in[2];
    const float* bias = (const float*)d_in[3];
    float* out = (float*)d_out;

    char* ws = (char*)d_ws;
    // ws layout: R (128*32*32 f32 = 512 KB) | W'_hi (32 MB bf16) | W'_lo (32 MB bf16)
    float* Rws = (float*)ws;
    const size_t R_BYTES  = (size_t)RANK * 32 * 32 * 4;
    const size_t WP_BYTES = (size_t)OUT_F * IN_F * 2;
    unsigned short* wph = (unsigned short*)(ws + R_BYTES);
    unsigned short* wpl = (unsigned short*)(ws + R_BYTES + WP_BYTES);

    oft_build_R<<<RANK, 1024, 0, stream>>>(oftw, Rws);
    oft_fold_W<<<dim3(OUT_F / 8, RANK), 256, 0, stream>>>(W, Rws, wph, wpl);
    oft_gemm<<<dim3(OUT_F / BN, M_TOK / BM), 256, 0, stream>>>(x, wph, wpl, bias, out);
}
